// QuantumAttention_59004260712621
// MI455X (gfx1250) — compile-verified, matrix operand order checked by reading, not run on hardware
//
#include <hip/hip_runtime.h>
#include <hip/hip_bf16.h>

// ---------------------------------------------------------------------------
// QuantumAttention for MI455X (gfx1250, wave32, WMMA bf16 w/ f32 accum)
//   B=1, S=4096, DIM=1024, H=16, HD=64
// Pipeline: cvt->bf16, 3x WMMA GEMM (QKV), fused flash-style attention with
// pointwise cos-modulation (keeps 1GB score matrix out of HBM), WMMA GEMM out.
// ---------------------------------------------------------------------------

#define S_LEN 4096
#define DIMSZ 1024
#define NHEAD 16
#define HDIM  64

typedef __attribute__((ext_vector_type(16))) __bf16       v16bf;
typedef __attribute__((ext_vector_type(8)))  float        v8f;
typedef __attribute__((ext_vector_type(8)))  unsigned int v8u;

// Load one 16-bit WMMA A/B fragment (16 bf16 per lane) from row-major storage.
// rowptr = pointer to this lane's row; koff = (lane>=16 ? 8 : 0).
// Per ISA 7.12.2: lanes 0-15 take K {0..7,16..23}, lanes 16-31 take {8..15,24..31}.
__device__ __forceinline__ v16bf load_frag16(const __bf16* rowptr, int koff) {
  const uint4 lo = *reinterpret_cast<const uint4*>(rowptr + koff);
  const uint4 hi = *reinterpret_cast<const uint4*>(rowptr + koff + 16);
  v8u u;
  u[0] = lo.x; u[1] = lo.y; u[2] = lo.z; u[3] = lo.w;
  u[4] = hi.x; u[5] = hi.y; u[6] = hi.z; u[7] = hi.w;
  return __builtin_bit_cast(v16bf, u);
}

__device__ __forceinline__ v8f wmma_bf16(v16bf a, v16bf b, v8f c) {
  // D(16x16 f32) = A(16x32 bf16) * B(32x16 bf16) + C
  return __builtin_amdgcn_wmma_f32_16x16x32_bf16(
      /*neg_a=*/false, a, /*neg_b=*/false, b,
      /*c_mod=*/(short)0, c, /*reuse_a=*/false, /*reuse_b=*/false);
}

// ---------------------------------------------------------------------------
// fp32 -> bf16 conversion
// ---------------------------------------------------------------------------
__global__ void cvt_f32_to_bf16(const float* __restrict__ in,
                                __bf16* __restrict__ out, int n) {
  int i = blockIdx.x * blockDim.x + threadIdx.x;
  if (i < n) out[i] = (__bf16)in[i];
}

// ---------------------------------------------------------------------------
// Y = A * W^T + bias.  A:[M][K] bf16, W:[N][K] bf16, bias:[N] f32.
// mode 0: store bf16 to [NHEAD][M][HDIM] (head-major, for attention)
// mode 1: store f32 to [M][N] (final output)
// Block: 256 thr = 8 waves, tile 128M x 128N, K-step 32. Wave = 32M x 64N.
// 8 WMMAs per wave per k-step (2 A-frags x 4 B-frags).
// ---------------------------------------------------------------------------
__global__ __launch_bounds__(256) void gemm_bf16_nt(
    const __bf16* __restrict__ A, const __bf16* __restrict__ W,
    const float* __restrict__ bias, void* __restrict__ outp,
    int M, int N, int K, int mode) {
  __shared__ __bf16 As[128][32];
  __shared__ __bf16 Bs[128][32];

  const int tid  = threadIdx.x;
  const int lane = tid & 31;
  const int wave = tid >> 5;
  const int wm   = wave >> 1;   // 0..3 -> 32 rows each
  const int wn   = wave & 1;    // 0..1 -> 64 cols each
  const int mb   = blockIdx.y * 128;
  const int nb   = blockIdx.x * 128;
  const int l15  = lane & 15;
  const int half = lane >> 4;
  const int koff = half * 8;

  v8f acc[2][4] = {};

  const int crow = tid >> 1;        // 0..127
  const int ck0  = (tid & 1) * 16;  // 0 or 16

  for (int kk = 0; kk < K; kk += 32) {
    __syncthreads();
    {
      const __bf16* src = A + (size_t)(mb + crow) * K + kk + ck0;
      *reinterpret_cast<uint4*>(&As[crow][ck0])     = *reinterpret_cast<const uint4*>(src);
      *reinterpret_cast<uint4*>(&As[crow][ck0 + 8]) = *reinterpret_cast<const uint4*>(src + 8);
    }
    {
      const __bf16* src = W + (size_t)(nb + crow) * K + kk + ck0;
      *reinterpret_cast<uint4*>(&Bs[crow][ck0])     = *reinterpret_cast<const uint4*>(src);
      *reinterpret_cast<uint4*>(&Bs[crow][ck0 + 8]) = *reinterpret_cast<const uint4*>(src + 8);
    }
    __syncthreads();

    v16bf af[2], bf[4];
    for (int mt = 0; mt < 2; ++mt)
      af[mt] = load_frag16(&As[wm * 32 + mt * 16 + l15][0], koff);
    for (int nt = 0; nt < 4; ++nt)
      bf[nt] = load_frag16(&Bs[wn * 64 + nt * 16 + l15][0], koff);
    for (int mt = 0; mt < 2; ++mt)
      for (int nt = 0; nt < 4; ++nt)
        acc[mt][nt] = wmma_bf16(af[mt], bf[nt], acc[mt][nt]);
  }

  for (int mt = 0; mt < 2; ++mt) {
    for (int nt = 0; nt < 4; ++nt) {
      for (int r = 0; r < 8; ++r) {
        const int m = mb + wm * 32 + mt * 16 + r + half * 8;
        const int n = nb + wn * 64 + nt * 16 + l15;
        const float v = acc[mt][nt][r] + bias[n];
        if (mode == 0) {
          __bf16* ob = (__bf16*)outp;          // [NHEAD][M][HDIM]
          const int h = n >> 6, hd = n & 63;
          ob[((size_t)h * M + m) * HDIM + hd] = (__bf16)v;
        } else {
          float* of = (float*)outp;            // [M][N]
          of[(size_t)m * N + n] = v;
        }
      }
    }
  }
}

// ---------------------------------------------------------------------------
// Fused flash-style attention with quantum-interference modulation.
// Grid: (S/128, NHEAD). Block: 256 thr = 8 waves; wave owns 16 query rows.
// 64-key tiles double-buffered through registers; modulation is pointwise so
// online softmax is exact. O:[S][DIM] bf16.
// ---------------------------------------------------------------------------
__global__ __launch_bounds__(256) void quantum_attn(
    const __bf16* __restrict__ Q, const __bf16* __restrict__ Km,
    const __bf16* __restrict__ V, const float* __restrict__ phase,
    __bf16* __restrict__ O) {
  __shared__ __bf16 Ks[64][HDIM];     // [key][hd]  -> B frag for scores   (8KB)
  __shared__ __bf16 Vs[HDIM][64];     // [hd][key]  -> B frag for P*V      (8KB)
  __shared__ __bf16 Pl[8][16][64];    // per-wave P tile relayout buffer  (16KB)

  const int tid  = threadIdx.x;
  const int lane = tid & 31;
  const int wave = tid >> 5;
  const int l15  = lane & 15;
  const int half = lane >> 4;
  const int koff = half * 8;
  const int h    = blockIdx.y;
  const int qw   = blockIdx.x * 128 + wave * 16;

  const float ph    = phase[h];
  const float scale = 0.125f;  // 1/sqrt(64)

  // Q fragments: two K-steps of 32 covering HD=64 (registers for whole pass)
  const __bf16* qrow = Q + ((size_t)h * S_LEN + qw + l15) * HDIM;
  v16bf qf0 = load_frag16(qrow, koff);
  v16bf qf1 = load_frag16(qrow + 32, koff);

  float mrow[8], lrow[8];
  v8f o[4] = {};
  for (int r = 0; r < 8; ++r) { mrow[r] = -1e30f; lrow[r] = 0.f; }

  // cooperative tile loads: each thread moves 32B of K and 32B of V
  const int ckey = tid >> 2;          // 0..63
  const int chd0 = (tid & 3) * 16;    // 0,16,32,48

  uint4 kr0, kr1, vr0, vr1;           // staged next tile (double buffer in regs)
  {
    const __bf16* ksrc = Km + ((size_t)h * S_LEN + ckey) * HDIM + chd0;
    kr0 = *reinterpret_cast<const uint4*>(ksrc);
    kr1 = *reinterpret_cast<const uint4*>(ksrc + 8);
    const __bf16* vsrc = V + ((size_t)h * S_LEN + ckey) * HDIM + chd0;
    vr0 = *reinterpret_cast<const uint4*>(vsrc);
    vr1 = *reinterpret_cast<const uint4*>(vsrc + 8);
  }

  for (int kt = 0; kt < S_LEN; kt += 64) {
    __syncthreads();   // previous iteration's LDS reads done
    // commit staged tile: K row-major, V transposed into [hd][key]
    *reinterpret_cast<uint4*>(&Ks[ckey][chd0])     = kr0;
    *reinterpret_cast<uint4*>(&Ks[ckey][chd0 + 8]) = kr1;
    {
      union { uint4 q[2]; __bf16 b[16]; } cv;
      cv.q[0] = vr0; cv.q[1] = vr1;
      for (int j = 0; j < 16; ++j) Vs[chd0 + j][ckey] = cv.b[j];
    }
    __syncthreads();

    // stage next tile while this one is being consumed
    if (kt + 64 < S_LEN) {
      const __bf16* ksrc = Km + ((size_t)h * S_LEN + kt + 64 + ckey) * HDIM + chd0;
      kr0 = *reinterpret_cast<const uint4*>(ksrc);
      kr1 = *reinterpret_cast<const uint4*>(ksrc + 8);
      const __bf16* vsrc = V + ((size_t)h * S_LEN + kt + 64 + ckey) * HDIM + chd0;
      vr0 = *reinterpret_cast<const uint4*>(vsrc);
      vr1 = *reinterpret_cast<const uint4*>(vsrc + 8);
    }

    // --- scores S = Q K^T * scale, then quantum modulation (4 x 16 keys) ---
    float sm[4][8];
    for (int nt = 0; nt < 4; ++nt) {
      v8f s = {};
      v16bf b0 = load_frag16(&Ks[nt * 16 + l15][0],  koff);
      v16bf b1 = load_frag16(&Ks[nt * 16 + l15][32], koff);
      s = wmma_bf16(qf0, b0, s);
      s = wmma_bf16(qf1, b1, s);
      for (int r = 0; r < 8; ++r) {
        float sc = s[r] * scale;
        sm[nt][r] = sc * (1.f + 0.1f * __cosf(ph * sc));
      }
    }

    // --- online softmax (row = lanes of one half-wave) ---
    float alpha[8];
    for (int r = 0; r < 8; ++r) {
      float mx = fmaxf(fmaxf(sm[0][r], sm[1][r]), fmaxf(sm[2][r], sm[3][r]));
      mx = fmaxf(mx, __shfl_xor(mx, 1, 32));
      mx = fmaxf(mx, __shfl_xor(mx, 2, 32));
      mx = fmaxf(mx, __shfl_xor(mx, 4, 32));
      mx = fmaxf(mx, __shfl_xor(mx, 8, 32));
      const float mnew = fmaxf(mrow[r], mx);
      const float a  = __expf(mrow[r] - mnew);
      const float p0 = __expf(sm[0][r] - mnew);
      const float p1 = __expf(sm[1][r] - mnew);
      const float p2 = __expf(sm[2][r] - mnew);
      const float p3 = __expf(sm[3][r] - mnew);
      float rs = (p0 + p1) + (p2 + p3);
      rs += __shfl_xor(rs, 1, 32);
      rs += __shfl_xor(rs, 2, 32);
      rs += __shfl_xor(rs, 4, 32);
      rs += __shfl_xor(rs, 8, 32);
      lrow[r] = lrow[r] * a + rs;
      mrow[r] = mnew;
      alpha[r] = a;
      sm[0][r] = p0; sm[1][r] = p1; sm[2][r] = p2; sm[3][r] = p3;
    }
    for (int n4 = 0; n4 < 4; ++n4)
      for (int r = 0; r < 8; ++r) o[n4][r] *= alpha[r];

    // --- relayout P (C-layout) -> A-fragments via per-wave LDS tile ---
    for (int nt = 0; nt < 4; ++nt)
      for (int r = 0; r < 8; ++r)
        Pl[wave][r + half * 8][nt * 16 + l15] = (__bf16)sm[nt][r];
    v16bf pf0 = load_frag16(&Pl[wave][l15][0],  koff);
    v16bf pf1 = load_frag16(&Pl[wave][l15][32], koff);

    // --- O += P * V  (K-dim = 64 keys, two k-steps) ---
    for (int n4 = 0; n4 < 4; ++n4) {
      v16bf vb0 = load_frag16(&Vs[n4 * 16 + l15][0],  koff);
      v16bf vb1 = load_frag16(&Vs[n4 * 16 + l15][32], koff);
      o[n4] = wmma_bf16(pf0, vb0, o[n4]);
      o[n4] = wmma_bf16(pf1, vb1, o[n4]);
    }
  }

  // normalize and store O as bf16 [S][DIM]
  for (int n4 = 0; n4 < 4; ++n4) {
    for (int r = 0; r < 8; ++r) {
      const int q  = qw + r + half * 8;
      const int hd = n4 * 16 + l15;
      O[(size_t)q * DIMSZ + h * HDIM + hd] = (__bf16)(o[n4][r] / lrow[r]);
    }
  }
}

// ---------------------------------------------------------------------------
extern "C" void kernel_launch(void* const* d_in, const int* in_sizes, int n_in,
                              void* d_out, int out_size, void* d_ws, size_t ws_size,
                              hipStream_t stream) {
  const float* x     = (const float*)d_in[0];
  const float* wq    = (const float*)d_in[1];
  const float* bq    = (const float*)d_in[2];
  const float* wk    = (const float*)d_in[3];
  const float* bk    = (const float*)d_in[4];
  const float* wv    = (const float*)d_in[5];
  const float* bv    = (const float*)d_in[6];
  const float* wo    = (const float*)d_in[7];
  const float* bo    = (const float*)d_in[8];
  const float* phase = (const float*)d_in[9];
  float* out = (float*)d_out;

  char* ws = (char*)d_ws;
  // workspace layout (bytes), total 48 MiB
  __bf16* xb  = (__bf16*)(ws + 0);          // 4096x1024 bf16 : 8 MiB
  __bf16* wqb = (__bf16*)(ws + 8388608);    // 1024x1024 bf16 : 2 MiB
  __bf16* wkb = (__bf16*)(ws + 10485760);
  __bf16* wvb = (__bf16*)(ws + 12582912);
  __bf16* wob = (__bf16*)(ws + 14680064);
  __bf16* Qb  = (__bf16*)(ws + 16777216);   // [16][4096][64] : 8 MiB
  __bf16* Kb  = (__bf16*)(ws + 25165824);
  __bf16* Vb  = (__bf16*)(ws + 33554432);
  __bf16* Ab  = (__bf16*)(ws + 41943040);   // [4096][1024]   : 8 MiB

  const int NX = S_LEN * DIMSZ;   // 4194304
  const int NW = DIMSZ * DIMSZ;   // 1048576
  cvt_f32_to_bf16<<<(NX + 255) / 256, 256, 0, stream>>>(x,  xb,  NX);
  cvt_f32_to_bf16<<<(NW + 255) / 256, 256, 0, stream>>>(wq, wqb, NW);
  cvt_f32_to_bf16<<<(NW + 255) / 256, 256, 0, stream>>>(wk, wkb, NW);
  cvt_f32_to_bf16<<<(NW + 255) / 256, 256, 0, stream>>>(wv, wvb, NW);
  cvt_f32_to_bf16<<<(NW + 255) / 256, 256, 0, stream>>>(wo, wob, NW);

  dim3 gg(DIMSZ / 128, S_LEN / 128);  // (8, 32)
  gemm_bf16_nt<<<gg, 256, 0, stream>>>(xb, wqb, bq, (void*)Qb, S_LEN, DIMSZ, DIMSZ, 0);
  gemm_bf16_nt<<<gg, 256, 0, stream>>>(xb, wkb, bk, (void*)Kb, S_LEN, DIMSZ, DIMSZ, 0);
  gemm_bf16_nt<<<gg, 256, 0, stream>>>(xb, wvb, bv, (void*)Vb, S_LEN, DIMSZ, DIMSZ, 0);

  dim3 ga(S_LEN / 128, NHEAD);        // (32, 16)
  quantum_attn<<<ga, 256, 0, stream>>>(Qb, Kb, Vb, phase, Ab);

  gemm_bf16_nt<<<gg, 256, 0, stream>>>(Ab, wob, bo, (void*)out, S_LEN, DIMSZ, DIMSZ, 1);
}
